// LocalRelativeAttentionBlock_31490700214934
// MI455X (gfx1250) — compile-verified, matrix operand order checked by reading, not run on hardware
//
#include <hip/hip_runtime.h>
#include <hip/hip_bf16.h>

typedef __attribute__((ext_vector_type(16))) _Float16 v16h;
typedef __attribute__((ext_vector_type(8)))  _Float16 v8h;
typedef __attribute__((ext_vector_type(8)))  float    v8f;

#define WMMA_F16(a, b, c) \
  __builtin_amdgcn_wmma_f32_16x16x32_f16(false, (a), false, (b), (short)0, (c), false, false)

// ---------------------------------------------------------------------------
// Problem constants
// ---------------------------------------------------------------------------
constexpr int  Bc   = 4;
constexpr int  Sc   = 8192;
constexpr int  Ec   = 256;   // embedding, K of all GEMMs
constexpr int  Hc   = 8;
constexpr int  DKc  = 64;
constexpr int  DVc  = 32;
constexpr int  LBc  = 32;
constexpr int  WINc = 65;
constexpr long Mrows = (long)Bc * Sc;   // 32768

// ---------------------------------------------------------------------------
// Fragment loaders.
// A fragment (16x32 f16): lane L -> row (L&15); K chunk 0..7 at k0+(L>>4)*8,
// chunk 8..15 at +16 further.  B fragments are identical because weights are
// pre-transposed to [N][K] (column of B == contiguous K run).
// ---------------------------------------------------------------------------
static __device__ __forceinline__ v16h ld2chunks(const _Float16* p) {
  v8h lo = *(const v8h*)(p);
  v8h hi = *(const v8h*)(p + 16);
  v16h r;
#pragma unroll
  for (int i = 0; i < 8; ++i) { r[i] = lo[i]; r[8 + i] = hi[i]; }
  return r;
}

static __device__ __forceinline__ v16h ldfrag(const _Float16* base, int ld, int k0) {
  const int l = threadIdx.x & 31;
  return ld2chunks(base + (size_t)(l & 15) * ld + k0 + (l >> 4) * 8);
}

// ---------------------------------------------------------------------------
// Conversions
// ---------------------------------------------------------------------------
__global__ void k_cvt_f16(const float* __restrict__ s, _Float16* __restrict__ d, long n) {
  long i = (long)blockIdx.x * blockDim.x + threadIdx.x;
  if (i < n) d[i] = (_Float16)s[i];
}

// WT[n*K + k] = W[k*N + n]  (f32 -> f16, transpose to N-major)
__global__ void k_tcvt(const float* __restrict__ W, _Float16* __restrict__ WT, int K, int N) {
  long i = (long)blockIdx.x * blockDim.x + threadIdx.x;
  if (i >= (long)K * N) return;
  int k = (int)(i / N), n = (int)(i % N);
  WT[(size_t)n * K + k] = (_Float16)W[i];
}

// ---------------------------------------------------------------------------
// Generic WMMA GEMM:  C[M x N] = act( A[M x 256] * BT[N x 256]^T + bias )
// Block: 256 threads = 8 waves; block tile 64x64; wave tile 16x32 (2 acc).
// ---------------------------------------------------------------------------
__global__ __launch_bounds__(256)
void k_gemm(const _Float16* __restrict__ A, const _Float16* __restrict__ BT,
            const float* __restrict__ bias, _Float16* __restrict__ outh,
            float* __restrict__ outf, int N, int relu) {
  const int l   = threadIdx.x & 31;
  const int hs  = l >> 4;
  const int idx = l & 15;
  const int wid = threadIdx.x >> 5;
  const int row0 = blockIdx.x * 64 + (wid & 3) * 16;
  const int col0 = blockIdx.y * 64 + (wid >> 2) * 32;

  const _Float16* Ab = A  + (size_t)row0 * Ec;
  const _Float16* B0 = BT + (size_t)col0 * Ec;
  const _Float16* B1 = BT + (size_t)(col0 + 16) * Ec;

  v8f acc0 = {}; v8f acc1 = {};
#pragma unroll
  for (int k0 = 0; k0 < Ec; k0 += 32) {
    v16h a  = ldfrag(Ab, Ec, k0);
    v16h b0 = ldfrag(B0, Ec, k0);
    v16h b1 = ldfrag(B1, Ec, k0);
    acc0 = WMMA_F16(a, b0, acc0);
    acc1 = WMMA_F16(a, b1, acc1);
  }

#pragma unroll
  for (int r = 0; r < 8; ++r) {
    const int row = row0 + r + 8 * hs;
    const int c0  = col0 + idx;
    const int c1  = col0 + 16 + idx;
    float v0 = acc0[r] + bias[c0];
    float v1 = acc1[r] + bias[c1];
    if (relu) { v0 = fmaxf(v0, 0.f); v1 = fmaxf(v1, 0.f); }
    if (outh) { outh[(size_t)row * N + c0] = (_Float16)v0;
                outh[(size_t)row * N + c1] = (_Float16)v1; }
    if (outf) { outf[(size_t)row * N + c0] = v0;
                outf[(size_t)row * N + c1] = v1; }
  }
}

// ---------------------------------------------------------------------------
// Banded local attention, one wave per (16-query tile, head, batch).
// scores[16x80] = Q[16x64] x Kwin^T[64x80]  (5 groups x 2 chained WMMAs)
// softmax (fp32, rel_bias + band + seq mask, deferred 1/sum)
// out[16x32]   = P[16x96] x Vwin[96x32]     (2 tiles x 3 chained WMMAs)
// ---------------------------------------------------------------------------
__global__ __launch_bounds__(32)
void k_attn(const _Float16* __restrict__ Qh, const _Float16* __restrict__ Kh,
            const _Float16* __restrict__ Vh, const float* __restrict__ relb,
            _Float16* __restrict__ AttnH) {
  __shared__ __align__(16) float     sc[16][80];
  __shared__ __align__(16) _Float16  pb[16][96];
  __shared__ __align__(16) _Float16  vt[32][96];
  __shared__ float ssum[16];

  const int l   = threadIdx.x;
  const int hs  = l >> 4;
  const int idx = l & 15;
  const int tile0 = blockIdx.x * 16;
  const int h     = blockIdx.y;
  const int b     = blockIdx.z;
  const size_t qrow = (size_t)b * Sc + tile0;

  // Q fragments: 16x64 over two K chunks
  const _Float16* qb = Qh + qrow * (Hc * DKc) + h * DKc;
  const v16h aq0 = ldfrag(qb, Hc * DKc, 0);
  const v16h aq1 = ldfrag(qb, Hc * DKc, 32);

  // ---- scores ----
  for (int g = 0; g < 5; ++g) {
    const int j  = g * 16 + idx;                      // key index in 80-window
    int kp = tile0 - LBc + j;                         // key position (may be OOB)
    int kpc = kp < 0 ? 0 : (kp > Sc - 1 ? Sc - 1 : kp);   // clamp (masked later)
    const _Float16* kb = Kh + ((size_t)b * Sc + kpc) * (Hc * DKc) + h * DKc;
    v16h b0 = ld2chunks(kb + 0  + hs * 8);
    v16h b1 = ld2chunks(kb + 32 + hs * 8);
    v8f acc = {};
    acc = WMMA_F16(aq0, b0, acc);
    acc = WMMA_F16(aq1, b1, acc);
#pragma unroll
    for (int r = 0; r < 8; ++r) sc[r + 8 * hs][g * 16 + idx] = acc[r];
  }

  // zero P buffer (invalid/padding entries stay 0)
  {
    float* pf = (float*)(&pb[0][0]);
    for (int i = l; i < 16 * 96 / 2; i += 32) pf[i] = 0.f;
  }
  __syncthreads();

  // ---- V window, transposed into LDS: vt[vdim][key] ----
  // Vectorized: each key's 32 values are one contiguous 64B run -> 4x b128.
  for (int j = l; j < 96; j += 32) {
    const int kp = tile0 - LBc + j;
    if (j < 80 && kp >= 0 && kp < Sc) {
      const v8h* vp = (const v8h*)(Vh + ((size_t)b * Sc + kp) * (Hc * DVc) + h * DVc);
      const v8h v0 = vp[0], v1 = vp[1], v2 = vp[2], v3 = vp[3];
#pragma unroll
      for (int i = 0; i < 8; ++i) {
        vt[i][j]      = v0[i];
        vt[8 + i][j]  = v1[i];
        vt[16 + i][j] = v2[i];
        vt[24 + i][j] = v3[i];
      }
    } else {
#pragma unroll
      for (int vd = 0; vd < DVc; ++vd) vt[vd][j] = (_Float16)0.f;
    }
  }
  __syncthreads();

  // ---- softmax: lane m handles query row m; keys j in [m, m+64] ----
  if (l < 16) {
    const int m = l;
    const float scale = 0.125f;                        // 1/sqrt(64)
    float mx = -1e30f;
    for (int w = 0; w < WINc; ++w) {
      const int j  = m + w;
      const int kp = tile0 - LBc + j;
      if (kp >= 0 && kp < Sc) {
        const float s = sc[m][j] * scale + relb[h * WINc + w];
        sc[m][j] = s;
        mx = fmaxf(mx, s);
      }
    }
    float sum = 0.f;
    for (int w = 0; w < WINc; ++w) {
      const int j  = m + w;
      const int kp = tile0 - LBc + j;
      if (kp >= 0 && kp < Sc) {
        const float e = __expf(sc[m][j] - mx);
        sum += e;
        pb[m][j] = (_Float16)e;
      }
    }
    ssum[m] = 1.f / sum;
  }
  __syncthreads();

  // ---- P x V ----
#pragma unroll
  for (int n0 = 0; n0 < DVc; n0 += 16) {
    v8f acc = {};
#pragma unroll
    for (int kg = 0; kg < 3; ++kg) {
      v16h ap = ld2chunks(&pb[idx][kg * 32 + hs * 8]);
      v16h bp = ld2chunks(&vt[n0 + idx][kg * 32 + hs * 8]);
      acc = WMMA_F16(ap, bp, acc);
    }
#pragma unroll
    for (int r = 0; r < 8; ++r) {
      const int row = r + 8 * hs;
      const float v = acc[r] * ssum[row];
      AttnH[(qrow + row) * (Hc * DVc) + h * DVc + n0 + idx] = (_Float16)v;
    }
  }
}

// ---------------------------------------------------------------------------
// y = LayerNorm(A + R) * g + beta ; optional f16 copy.  One wave per row.
// ---------------------------------------------------------------------------
__global__ __launch_bounds__(256)
void k_addln(const float* __restrict__ A, const float* __restrict__ R,
             const float* __restrict__ g, const float* __restrict__ be,
             float* __restrict__ y, _Float16* __restrict__ yh) {
  const int l   = threadIdx.x & 31;
  const int wid = threadIdx.x >> 5;
  const size_t row = (size_t)blockIdx.x * 8 + wid;
  const float* ap = A + row * Ec;
  const float* rp = R + row * Ec;

  float v[8]; float s = 0.f, sq = 0.f;
#pragma unroll
  for (int i = 0; i < 8; ++i) {
    const int c = l + i * 32;
    const float x = ap[c] + rp[c];
    v[i] = x; s += x; sq += x * x;
  }
#pragma unroll
  for (int o = 16; o > 0; o >>= 1) {
    s  += __shfl_xor(s,  o, 32);
    sq += __shfl_xor(sq, o, 32);
  }
  const float mean = s * (1.f / Ec);
  const float var  = sq * (1.f / Ec) - mean * mean;
  const float rs   = rsqrtf(var + 1e-5f);
#pragma unroll
  for (int i = 0; i < 8; ++i) {
    const int c = l + i * 32;
    const float o = (v[i] - mean) * rs * g[c] + be[c];
    y[row * Ec + c] = o;
    if (yh) yh[row * Ec + c] = (_Float16)o;
  }
}

// ---------------------------------------------------------------------------
// Host launcher
// ---------------------------------------------------------------------------
extern "C" void kernel_launch(void* const* d_in, const int* in_sizes, int n_in,
                              void* d_out, int out_size, void* d_ws, size_t ws_size,
                              hipStream_t stream) {
  (void)in_sizes; (void)n_in; (void)out_size; (void)ws_size;

  const float* x     = (const float*)d_in[0];
  const float* Wq    = (const float*)d_in[1];
  const float* bq    = (const float*)d_in[2];
  const float* Wk    = (const float*)d_in[3];
  const float* bk    = (const float*)d_in[4];
  const float* Wv    = (const float*)d_in[5];
  const float* bv    = (const float*)d_in[6];
  const float* Wo    = (const float*)d_in[7];
  const float* bo    = (const float*)d_in[8];
  const float* relb  = (const float*)d_in[9];
  const float* ln1g  = (const float*)d_in[10];
  const float* ln1b  = (const float*)d_in[11];
  const float* Wlin  = (const float*)d_in[12];
  const float* blin  = (const float*)d_in[13];
  const float* ln2g  = (const float*)d_in[14];
  const float* ln2b  = (const float*)d_in[15];
  float* out = (float*)d_out;

  // workspace carve-up
  char* ws = (char*)d_ws;
  size_t off = 0;
  auto carve = [&](size_t bytes) { char* p = ws + off; off += (bytes + 255) & ~(size_t)255; return p; };
  _Float16* xh    = (_Float16*)carve(Mrows * Ec * 2);            // x as f16
  _Float16* Qh    = (_Float16*)carve(Mrows * Hc * DKc * 2);      // 32768x512
  _Float16* Khb   = (_Float16*)carve(Mrows * Hc * DKc * 2);
  _Float16* Vhb   = (_Float16*)carve(Mrows * Hc * DVc * 2);      // 32768x256
  _Float16* AttnH = (_Float16*)carve(Mrows * Ec * 2);
  float*    O1    = (float*)   carve(Mrows * Ec * 4);            // reused by FFN
  float*    xs32  = (float*)   carve(Mrows * Ec * 4);
  _Float16* xsh   = (_Float16*)carve(Mrows * Ec * 2);
  _Float16* WqT   = (_Float16*)carve((size_t)Ec * Hc * DKc * 2);
  _Float16* WkT   = (_Float16*)carve((size_t)Ec * Hc * DKc * 2);
  _Float16* WvT   = (_Float16*)carve((size_t)Ec * Ec * 2);
  _Float16* WoT   = (_Float16*)carve((size_t)Ec * Ec * 2);
  _Float16* WlinT = (_Float16*)carve((size_t)Ec * Ec * 2);

  // 1. conversions
  {
    long n = Mrows * Ec;
    k_cvt_f16<<<(unsigned)((n + 255) / 256), 256, 0, stream>>>(x, xh, n);
    long t = (long)Ec * (Hc * DKc);
    k_tcvt<<<(unsigned)((t + 255) / 256), 256, 0, stream>>>(Wq, WqT, Ec, Hc * DKc);
    k_tcvt<<<(unsigned)((t + 255) / 256), 256, 0, stream>>>(Wk, WkT, Ec, Hc * DKc);
    long t2 = (long)Ec * Ec;
    k_tcvt<<<(unsigned)((t2 + 255) / 256), 256, 0, stream>>>(Wv, WvT, Ec, Ec);
    k_tcvt<<<(unsigned)((t2 + 255) / 256), 256, 0, stream>>>(Wo, WoT, Ec, Ec);
    k_tcvt<<<(unsigned)((t2 + 255) / 256), 256, 0, stream>>>(Wlin, WlinT, Ec, Ec);
  }

  const unsigned gm = (unsigned)(Mrows / 64);   // 512

  // 2. Q/K/V projections
  k_gemm<<<dim3(gm, (Hc * DKc) / 64), 256, 0, stream>>>(xh, WqT, bq, Qh,  nullptr, Hc * DKc, 0);
  k_gemm<<<dim3(gm, (Hc * DKc) / 64), 256, 0, stream>>>(xh, WkT, bk, Khb, nullptr, Hc * DKc, 0);
  k_gemm<<<dim3(gm, Ec / 64),         256, 0, stream>>>(xh, WvT, bv, Vhb, nullptr, Ec, 0);

  // 3. banded attention
  k_attn<<<dim3(Sc / 16, Hc, Bc), 32, 0, stream>>>(Qh, Khb, Vhb, relb, AttnH);

  // 4. output projection -> O1 (f32)
  k_gemm<<<dim3(gm, Ec / 64), 256, 0, stream>>>(AttnH, WoT, bo, nullptr, O1, Ec, 0);

  // 5. xs = LN(x + O1)  -> xs32 (f32) + xsh (f16)
  k_addln<<<(unsigned)(Mrows / 8), 256, 0, stream>>>(x, O1, ln1g, ln1b, xs32, xsh);

  // 6. FFN: relu(xs @ W_lin + b) -> O1 (reuse)
  k_gemm<<<dim3(gm, Ec / 64), 256, 0, stream>>>(xsh, WlinT, blin, nullptr, O1, Ec, 1);

  // 7. out = LN(xs + O1)
  k_addln<<<(unsigned)(Mrows / 8), 256, 0, stream>>>(xs32, O1, ln2g, ln2b, out, nullptr);
}
